// MultiHeadAttention_15513421873209
// MI455X (gfx1250) — matrix operand order repaired and checked by reading, not compiled, not run on hardware
//
#include <hip/hip_runtime.h>

// ---------------------------------------------------------------------------
// Types
// ---------------------------------------------------------------------------
typedef __bf16 bf16_t;
typedef __attribute__((ext_vector_type(16))) __bf16 v16bf;
typedef __attribute__((ext_vector_type(8)))  __bf16 v8bf;
typedef __attribute__((ext_vector_type(8)))  float  v8f;
typedef __attribute__((ext_vector_type(4)))  float  v4f;

#define D_MODEL 1024
#define NH      16
#define DK      64
#define BQ      4
#define SEQ     2048
#define MTOT    (BQ * SEQ)   // 8192 rows

// ---------------------------------------------------------------------------
// CDNA5 async global->LDS copy (GLOBAL_LOAD_ASYNC_TO_LDS_B128, ASYNCcnt).
// LDS byte address = low 32 bits of the generic pointer (ISA 10.2: LDS
// aperture truncates to addr[31:0]).
// ---------------------------------------------------------------------------
__device__ __forceinline__ void async_load_b128(void* lds_dst,
                                                const void* gsrc) {
  const unsigned lds_off = (unsigned)(unsigned long long)lds_dst;
  asm volatile("global_load_async_to_lds_b128 %0, %1, off"
               :: "v"(lds_off), "v"(gsrc)
               : "memory");
}
__device__ __forceinline__ void wait_async0() {
  asm volatile("s_wait_asynccnt 0x0" ::: "memory");
}

// ---------------------------------------------------------------------------
// WMMA helpers (CDNA5 16x16x32 bf16, f32 accumulate)
// ---------------------------------------------------------------------------
__device__ __forceinline__ v8f wmma_bf16(v16bf a, v16bf b, v8f c) {
  return __builtin_amdgcn_wmma_f32_16x16x32_bf16(
      /*neg_a=*/false, a, /*neg_b=*/false, b,
      /*c_mod=*/(short)0, c, /*reuse_a=*/false, /*reuse_b=*/false);
}

__device__ __forceinline__ v16bf combine16(v8bf lo, v8bf hi) {
  v16bf r;
#pragma unroll
  for (int i = 0; i < 8; ++i) { r[i] = lo[i]; r[i + 8] = hi[i]; }
  return r;
}

// A-matrix 16x32 bf16 fragment (M=row0.., K=k0..k0+31) from LDS [M][K] rows.
// Lanes 0-15 row M=lane; halves h<8 -> K=k0+(lane>=16?8:0)+h, h>=8 -> +16.
__device__ __forceinline__ v16bf lds_a_frag(const bf16_t* base, int pitch,
                                            int row0, int k0) {
  const int lane = threadIdx.x & 31;
  const bf16_t* p = base + (size_t)(row0 + (lane & 15)) * pitch + k0 +
                    ((lane >> 4) << 3);
  return combine16(*(const v8bf*)p, *(const v8bf*)(p + 16));
}

// B-matrix 32x16 bf16 fragment (K=k0..k0+31, N=n0..n0+15) from LDS [N][K]
// rows: N=lane%16, halves h<8 -> K=k0+(lane>=16?8:0)+h, h>=8 -> +16.
__device__ __forceinline__ v16bf lds_b_frag(const bf16_t* base, int pitch,
                                            int n0, int k0) {
  const int lane = threadIdx.x & 31;
  const bf16_t* p = base + (size_t)(n0 + (lane & 15)) * pitch + k0 +
                    ((lane >> 4) << 3);
  return combine16(*(const v8bf*)p, *(const v8bf*)(p + 16));
}

__device__ __forceinline__ float half_reduce_max(float v) {
#pragma unroll
  for (int m = 1; m < 16; m <<= 1) v = fmaxf(v, __shfl_xor(v, m, 32));
  return v;
}
__device__ __forceinline__ float half_reduce_sum(float v) {
#pragma unroll
  for (int m = 1; m < 16; m <<= 1) v += __shfl_xor(v, m, 32);
  return v;
}

// ---------------------------------------------------------------------------
// GEMM: Y[M,1024] = X[M,1024] @ W[1024,1024]^T + bias   (torch Linear)
//   IN_BF16:  X is bf16 (async-copied straight into LDS);
//             else fp32 (converted to bf16 on LDS stage-in)
//   OUT_HEADSPLIT: write bf16 to [B*H, S, DK] (Q/K/V path, bias fused)
//                  else write fp32 [M, 1024] (final projection)
// Block: 128x128 tile, BK=32, 256 threads = 8 waves in 2x4, each wave 64x32.
// ---------------------------------------------------------------------------
template <bool IN_BF16, bool OUT_HEADSPLIT>
__global__ __launch_bounds__(256) void gemm_bias_kernel(
    const void* __restrict__ Xv, const float* __restrict__ W,
    const float* __restrict__ bias, void* __restrict__ Yv) {
  constexpr int BM = 128, BN = 128, BK = 32, LP = BK + 8;  // LDS pitch 40
  __shared__ bf16_t Al[BM * LP];
  __shared__ bf16_t Bl[BN * LP];

  const int nblk = blockIdx.x * BN;
  const int mblk = blockIdx.y * BM;
  const int tid  = threadIdx.x;
  const int lane = tid & 31;
  const int wave = tid >> 5;
  const int wm = wave >> 2;  // 0..1 -> 64-row half
  const int wn = wave & 3;   // 0..3 -> 32-col slice

  v8f acc[4][2];
#pragma unroll
  for (int mt = 0; mt < 4; ++mt)
#pragma unroll
    for (int nt = 0; nt < 2; ++nt)
#pragma unroll
      for (int i = 0; i < 8; ++i) acc[mt][nt][i] = 0.0f;

  const int lrow = tid >> 1;          // 0..127
  const int lcol = (tid & 1) * 16;    // 0 or 16

  for (int k0 = 0; k0 < D_MODEL; k0 += BK) {
    __syncthreads();
    // ---- stage A tile ----
    if (IN_BF16) {
      const bf16_t* X = (const bf16_t*)Xv;
      const bf16_t* src = X + (size_t)(mblk + lrow) * D_MODEL + k0 + lcol;
      // async copy straight into LDS (2 x 16B per thread), ASYNCcnt-tracked
      async_load_b128(&Al[lrow * LP + lcol], src);
      async_load_b128(&Al[lrow * LP + lcol + 8], src + 8);
    } else {
      const float* X = (const float*)Xv;
      const float* src = X + (size_t)(mblk + lrow) * D_MODEL + k0 + lcol;
#pragma unroll
      for (int c = 0; c < 16; c += 4) {
        v4f x = *(const v4f*)(src + c);
        bf16_t* dst = &Al[lrow * LP + lcol + c];
        dst[0] = (bf16_t)x.x; dst[1] = (bf16_t)x.y;
        dst[2] = (bf16_t)x.z; dst[3] = (bf16_t)x.w;
      }
    }
    // ---- stage W tile (W is [N,K] row-major fp32 -> bf16) ----
    {
      const float* src = W + (size_t)(nblk + lrow) * D_MODEL + k0 + lcol;
#pragma unroll
      for (int c = 0; c < 16; c += 4) {
        v4f x = *(const v4f*)(src + c);
        bf16_t* dst = &Bl[lrow * LP + lcol + c];
        dst[0] = (bf16_t)x.x; dst[1] = (bf16_t)x.y;
        dst[2] = (bf16_t)x.z; dst[3] = (bf16_t)x.w;
      }
      if (k0 + BK < D_MODEL) {  // hint next tiles -> global_prefetch_b8
        __builtin_prefetch(src + BK, 0, 1);
        if (!IN_BF16)
          __builtin_prefetch((const float*)Xv +
                                 (size_t)(mblk + lrow) * D_MODEL + k0 + BK,
                             0, 1);
      }
    }
    if (IN_BF16) wait_async0();  // drain own async copies before barrier
    __syncthreads();

    v16bf a[4], b[2];
#pragma unroll
    for (int mt = 0; mt < 4; ++mt)
      a[mt] = lds_a_frag(Al, LP, wm * 64 + mt * 16, 0);
#pragma unroll
    for (int nt = 0; nt < 2; ++nt)
      b[nt] = lds_b_frag(Bl, LP, wn * 32 + nt * 16, 0);
#pragma unroll
    for (int mt = 0; mt < 4; ++mt)
#pragma unroll
      for (int nt = 0; nt < 2; ++nt)
        acc[mt][nt] = wmma_bf16(a[mt], b[nt], acc[mt][nt]);
  }

  // ---- epilogue: bias add + store ----
#pragma unroll
  for (int nt = 0; nt < 2; ++nt) {
    const int ncol = nblk + wn * 32 + nt * 16 + (lane & 15);
    const float bn = bias[ncol];
#pragma unroll
    for (int mt = 0; mt < 4; ++mt) {
#pragma unroll
      for (int r = 0; r < 8; ++r) {
        const int mrow = mblk + wm * 64 + mt * 16 + ((lane < 16) ? r : 8 + r);
        const float val = acc[mt][nt][r] + bn;
        if (OUT_HEADSPLIT) {
          bf16_t* Y = (bf16_t*)Yv;
          const int bb = mrow >> 11, ss = mrow & 2047;
          const int hh = ncol >> 6, dd = ncol & 63;
          Y[(((size_t)(bb * NH + hh)) * SEQ + ss) * DK + dd] = (bf16_t)val;
        } else {
          float* Y = (float*)Yv;
          Y[(size_t)mrow * D_MODEL + ncol] = val;
        }
      }
    }
  }
}

// ---------------------------------------------------------------------------
// Causal flash attention over one head: Qh/Kh/Vh are bf16 [B*H, SEQ, DK].
// Block: 128 query rows (8 waves x 16 rows), key tiles of 64, online softmax.
// K tile staged via async global->LDS; V tile staged transposed by hand.
// Output: bf16 [B, S, D_MODEL] (heads re-concatenated) for final projection.
// ---------------------------------------------------------------------------
__global__ __launch_bounds__(256) void attn_kernel(
    const void* __restrict__ Qv, const void* __restrict__ Kv,
    const void* __restrict__ Vv, void* __restrict__ Ov) {
  constexpr int TQ = 128;        // query tile rows
  constexpr int TK = 64;         // key tile
  constexpr int KP = DK + 8;     // 72: Kt pitch  [key][d]
  constexpr int VP = TK + 8;     // 72: Vt/P pitch
  __shared__ bf16_t Kt[TK * KP];       // keys row-major   [key][d]
  __shared__ bf16_t Vt[DK * VP];       // values transposed [d][key]
  __shared__ bf16_t Pl[8 * 16 * VP];   // per-wave P tile  [16][TK]

  const bf16_t* Qh = (const bf16_t*)Qv;
  const bf16_t* Kh = (const bf16_t*)Kv;
  const bf16_t* Vh = (const bf16_t*)Vv;
  bf16_t* AttnO = (bf16_t*)Ov;

  const int qt   = blockIdx.x;   // 0..15
  const int bh   = blockIdx.y;   // 0..63
  const int tid  = threadIdx.x;
  const int lane = tid & 31;
  const int wave = tid >> 5;
  const size_t hbase = (size_t)bh * SEQ * DK;

  // Q fragments for this wave's 16 rows (held in registers for whole block)
  const int qrow  = qt * TQ + wave * 16 + (lane & 15);
  const int loff8 = (lane >> 4) << 3;  // A-frag lane K offset (0 / 8)
  v16bf qf[2];
#pragma unroll
  for (int kk = 0; kk < 2; ++kk) {
    const bf16_t* p = Qh + hbase + (size_t)qrow * DK + kk * 32 + loff8;
    qf[kk] = combine16(*(const v8bf*)p, *(const v8bf*)(p + 16));
  }

  float rowmax[8], rowsum[8];
  v8f outacc[4];
#pragma unroll
  for (int r = 0; r < 8; ++r) { rowmax[r] = -1e30f; rowsum[r] = 0.0f; }
#pragma unroll
  for (int dt = 0; dt < 4; ++dt)
#pragma unroll
    for (int i = 0; i < 8; ++i) outacc[dt][i] = 0.0f;

  const int nkt = (qt + 1) * (TQ / TK);  // causal: keys up to end of q tile
  for (int kt = 0; kt < nkt; ++kt) {
    __syncthreads();
    // ---- stage K tile (async global->LDS) and V tile (transposed) ----
    {
      const int key = tid >> 2;            // 0..63
      const int c0  = (tid & 3) * 16;      // 0/16/32/48
      const bf16_t* ks = Kh + hbase + (size_t)(kt * TK + key) * DK + c0;
      async_load_b128(&Kt[key * KP + c0], ks);
      async_load_b128(&Kt[key * KP + c0 + 8], ks + 8);
      const bf16_t* vs = Vh + hbase + (size_t)(kt * TK + key) * DK + c0;
      v8bf y0 = *(const v8bf*)vs;
      v8bf y1 = *(const v8bf*)(vs + 8);
#pragma unroll
      for (int i = 0; i < 8; ++i) {
        Vt[(c0 + i) * VP + key]     = y0[i];
        Vt[(c0 + 8 + i) * VP + key] = y1[i];
      }
    }
    wait_async0();  // K tile landed in LDS
    __syncthreads();

    // ---- scores S = (Q K^T) * 1/sqrt(dk), 16x64 per wave ----
    v8f sc[4];
#pragma unroll
    for (int nt = 0; nt < 4; ++nt) {
      v8f s;
#pragma unroll
      for (int i = 0; i < 8; ++i) s[i] = 0.0f;
      v16bf b0 = lds_b_frag(Kt, KP, nt * 16, 0);
      v16bf b1 = lds_b_frag(Kt, KP, nt * 16, 32);
      s = wmma_bf16(qf[0], b0, s);
      s = wmma_bf16(qf[1], b1, s);
      sc[nt] = s;
    }
    // scale + causal mask
#pragma unroll
    for (int nt = 0; nt < 4; ++nt) {
      const int colg = kt * TK + nt * 16 + (lane & 15);
#pragma unroll
      for (int r = 0; r < 8; ++r) {
        const int rowg = qt * TQ + wave * 16 + ((lane < 16) ? r : 8 + r);
        const float v = sc[nt][r] * 0.125f;  // 1/sqrt(64)
        sc[nt][r] = (colg > rowg) ? -1e30f : v;
      }
    }
    // ---- online softmax (rows live across 16-lane halves) ----
#pragma unroll
    for (int r = 0; r < 8; ++r) {
      float tmax = -1e30f;
#pragma unroll
      for (int nt = 0; nt < 4; ++nt) tmax = fmaxf(tmax, sc[nt][r]);
      tmax = half_reduce_max(tmax);
      const float mnew = fmaxf(rowmax[r], tmax);
      const float corr = __expf(rowmax[r] - mnew);
      rowmax[r] = mnew;
      float psum = 0.0f;
#pragma unroll
      for (int nt = 0; nt < 4; ++nt) {
        const float p = __expf(sc[nt][r] - mnew);
        sc[nt][r] = p;
        psum += p;
      }
      psum = half_reduce_sum(psum);
      rowsum[r] = rowsum[r] * corr + psum;
#pragma unroll
      for (int dt = 0; dt < 4; ++dt) outacc[dt][r] *= corr;
    }
    // ---- C-layout -> A-layout via wave-private LDS (bf16 P) ----
    bf16_t* Pw = &Pl[wave * 16 * VP];
#pragma unroll
    for (int nt = 0; nt < 4; ++nt) {
      const int col = nt * 16 + (lane & 15);
#pragma unroll
      for (int r = 0; r < 8; ++r) {
        const int row = (lane < 16) ? r : 8 + r;
        Pw[row * VP + col] = (bf16_t)sc[nt][r];
      }
    }
    // LDS is in-order per wave; make the cross-lane write->read explicit
    asm volatile("s_wait_dscnt 0" ::: "memory");
    // ---- O += P @ V ----
#pragma unroll
    for (int kk = 0; kk < 2; ++kk) {
      v16bf a = lds_a_frag(Pw, VP, 0, kk * 32);
#pragma unroll
      for (int dt = 0; dt < 4; ++dt) {
        v16bf b = lds_b_frag(Vt, VP, dt * 16, kk * 32);
        outacc[dt] = wmma_bf16(a, b, outacc[dt]);
      }
    }
  }

  // ---- normalize and store, heads re-concatenated: [B, S, H*DK] ----
  const int bb = bh >> 4, hh = bh & 15;
#pragma unroll
  for (int r = 0; r < 8; ++r) {
    const float inv = 1.0f / rowsum[r];
    const int rowl = wave * 16 + ((lane < 16) ? r : 8 + r);
    const int ss = qt * TQ + rowl;
#pragma unroll
    for (int dt = 0; dt < 4; ++dt) {
      const int dd = dt * 16 + (lane & 15);
      AttnO[((size_t)(bb * SEQ + ss)) * D_MODEL + hh * DK + dd] =
          (bf16_t)(outacc[dt][r] * inv);
    }
  }
}

// ---------------------------------------------------------------------------
// Launch
// ---------------------------------------------------------------------------
extern "C" void kernel_launch(void* const* d_in, const int* in_sizes, int n_in,
                              void* d_out, int out_size, void* d_ws,
                              size_t ws_size, hipStream_t stream) {
  (void)in_sizes; (void)n_in; (void)out_size; (void)ws_size;
  const float* q  = (const float*)d_in[0];
  const float* k  = (const float*)d_in[1];
  const float* v  = (const float*)d_in[2];
  // d_in[3] = mask (tril) -> causality implemented directly in attn_kernel
  const float* Wq = (const float*)d_in[4];
  const float* bq = (const float*)d_in[5];
  const float* Wk = (const float*)d_in[6];
  const float* bk = (const float*)d_in[7];
  const float* Wv = (const float*)d_in[8];
  const float* bv = (const float*)d_in[9];
  const float* Wo = (const float*)d_in[10];
  const float* bo = (const float*)d_in[11];

  const size_t headElems = (size_t)BQ * NH * SEQ * DK;  // 8.39M bf16 each
  bf16_t* Qh = (bf16_t*)d_ws;
  bf16_t* Kh = Qh + headElems;
  bf16_t* Vh = Kh + headElems;
  bf16_t* Ao = Vh + headElems;  // attention out, [B, S, D_MODEL] bf16

  dim3 gblk(D_MODEL / 128, MTOT / 128);  // (8, 64)
  gemm_bias_kernel<false, true><<<gblk, 256, 0, stream>>>(q, Wq, bq, Qh);
  gemm_bias_kernel<false, true><<<gblk, 256, 0, stream>>>(k, Wk, bk, Kh);
  gemm_bias_kernel<false, true><<<gblk, 256, 0, stream>>>(v, Wv, bv, Vh);

  attn_kernel<<<dim3(SEQ / 128, BQ * NH), 256, 0, stream>>>(Qh, Kh, Vh, Ao);

  gemm_bias_kernel<true, false><<<gblk, 256, 0, stream>>>(Ao, Wo, bo,
                                                          (float*)d_out);
}
